// StageNet_9706626089879
// MI455X (gfx1250) — matrix operand order repaired and checked by reading, not compiled, not run on hardware
//
#include <hip/hip_runtime.h>
#include <hip/hip_bf16.h>
#include <math.h>

typedef __attribute__((ext_vector_type(16))) _Float16 v16h;
typedef __attribute__((ext_vector_type(8)))  float    v8f;

#define Bc 1
#define Vc 5
#define Cc 16
#define Gc 4
#define Dc 32
#define Hc 192
#define Wc 256
#define HW (Hc*Wc)

// ds_swizzle bit-mode patterns: offset = xor<<10 | or<<5 | and ; and=0x1f keeps lanes
#define SWZ_X16 0x401f
#define SWZ_X8  0x201f
#define SWZ_X4  0x101f
#define SWZ_X2  0x081f
#define SWZ_X1  0x041f

#define FSWZ(v, pat) __int_as_float(__builtin_amdgcn_ds_swizzle(__float_as_int(v), pat))

__device__ __forceinline__ float wave_max32(float v) {
    v = fmaxf(v, FSWZ(v, SWZ_X16));
    v = fmaxf(v, FSWZ(v, SWZ_X8));
    v = fmaxf(v, FSWZ(v, SWZ_X4));
    v = fmaxf(v, FSWZ(v, SWZ_X2));
    v = fmaxf(v, FSWZ(v, SWZ_X1));
    return v;
}
__device__ __forceinline__ float wave_sum32(float v) {
    v += FSWZ(v, SWZ_X16);
    v += FSWZ(v, SWZ_X8);
    v += FSWZ(v, SWZ_X4);
    v += FSWZ(v, SWZ_X2);
    v += FSWZ(v, SWZ_X1);
    return v;
}
__device__ __forceinline__ float read_lane_f(float v, int lane_uniform) {
    return __int_as_float(__builtin_amdgcn_readlane(__float_as_int(v), lane_uniform));
}

// ---------------------------------------------------------------------------
// Precompute per-src-view rot(3x3)+trans(3) of  P_src @ inv(P_ref)  into ws.
// ---------------------------------------------------------------------------
__global__ void proj_precompute_kernel(const float* __restrict__ pm,
                                       float* __restrict__ ws) {
    if (threadIdx.x != 0 || blockIdx.x != 0) return;
    float P[Vc][16];
    for (int v = 0; v < Vc; ++v) {
        const float* E = pm + v * 32;        // proj_matrices[0, v, 0]
        const float* K = pm + v * 32 + 16;   // proj_matrices[0, v, 1]
        for (int i = 0; i < 3; ++i)
            for (int j = 0; j < 4; ++j) {
                float s = 0.f;
                for (int k = 0; k < 3; ++k) s += K[i * 4 + k] * E[k * 4 + j];
                P[v][i * 4 + j] = s;
            }
        for (int j = 0; j < 4; ++j) P[v][12 + j] = E[12 + j];  // [0,0,0,1]
    }
    const float* A = P[0];
    float a00 = A[0], a01 = A[1], a02 = A[2];
    float a10 = A[4], a11 = A[5], a12 = A[6];
    float a20 = A[8], a21 = A[9], a22 = A[10];
    float b0 = A[3], b1 = A[7], b2 = A[11];
    float c00 =  (a11 * a22 - a12 * a21);
    float c01 = -(a01 * a22 - a02 * a21);
    float c02 =  (a01 * a12 - a02 * a11);
    float c10 = -(a10 * a22 - a12 * a20);
    float c11 =  (a00 * a22 - a02 * a20);
    float c12 = -(a00 * a12 - a02 * a10);
    float c20 =  (a10 * a21 - a11 * a20);
    float c21 = -(a00 * a21 - a01 * a20);
    float c22 =  (a00 * a11 - a01 * a10);
    float det = a00 * c00 + a01 * c10 + a02 * c20;
    float id = 1.0f / det;
    float Ai[9] = { c00 * id, c01 * id, c02 * id,
                    c10 * id, c11 * id, c12 * id,
                    c20 * id, c21 * id, c22 * id };
    float ib0 = -(Ai[0] * b0 + Ai[1] * b1 + Ai[2] * b2);
    float ib1 = -(Ai[3] * b0 + Ai[4] * b1 + Ai[5] * b2);
    float ib2 = -(Ai[6] * b0 + Ai[7] * b1 + Ai[8] * b2);
    for (int s = 1; s < Vc; ++s) {
        const float* Ms = P[s];
        float* o = ws + (s - 1) * 12;
        for (int i = 0; i < 3; ++i) {
            for (int j = 0; j < 3; ++j)
                o[i * 3 + j] = Ms[i * 4 + 0] * Ai[0 * 3 + j] +
                               Ms[i * 4 + 1] * Ai[1 * 3 + j] +
                               Ms[i * 4 + 2] * Ai[2 * 3 + j];
            o[9 + i] = Ms[i * 4 + 0] * ib0 + Ms[i * 4 + 1] * ib1 +
                       Ms[i * 4 + 2] * ib2 + Ms[i * 4 + 3];
        }
    }
}

// ---------------------------------------------------------------------------
// Main kernel: one wave32 per pixel, lane = depth hypothesis (D == 32).
// Correlation via v_wmma_f32_16x16x32_f16:
//   A(g, c) = 0.25 * ref[c] * [c in group g]   (rows 4..15 and K 16..31 zero)
//   B(c, n) = f16(warped[depth n][c])          (lanes 16..31 / K>=16 zero)
// Result extraction is predication-proof:  cor[g] = D1[g] + swap16(lo?D2[g]:0)
// (valid because rows 4..15 of D are exact zeros), so every ds_swizzle is
// consumed by all lanes and stays unconditional.
// ---------------------------------------------------------------------------
__global__ __launch_bounds__(256) void stagenet_kernel(
    const float* __restrict__ feat,   // (V,B,C,H,W)
    const float* __restrict__ dh,     // (B,D,H,W)
    const float* __restrict__ rw,     // (G)
    const float* __restrict__ dir,    // (1)
    const float* __restrict__ RT,     // (4,12) rot+trans per src view
    float* __restrict__ out)          // depth|conf|prob|invmin|invmax
{
    const int lane = threadIdx.x & 31;
    const int wid  = threadIdx.x >> 5;
    const int pix  = blockIdx.x * 8 + wid;
    const int y = pix / Wc;
    const int x = pix - y * Wc;
    const bool lo = (lane < 16);

    // ---- reference features for this pixel (same for all lanes) ----
    float ref[16];
    const float* rp = feat + (size_t)y * Wc + x;          // v=0, c=0
    #pragma unroll
    for (int c = 0; c < 16; ++c) ref[c] = rp[(size_t)c * HW];

    // ---- this lane's depth hypothesis (lane = depth) ----
    const float depth = dh[(size_t)lane * HW + (size_t)y * Wc + x];

    // ---- build A matrix (constant across views) ----
    v16h Amat;
    {
        const int hl = lane >> 4;
        const int Lm = lane & 15;
        #pragma unroll
        for (int i = 0; i < 16; ++i) {
            float v = 0.0f;
            if (!hl) {
                float rv = ref[i];
                bool match = (Lm == (i >> 2));
                v = match ? 0.25f * rv : 0.0f;
            }
            Amat[i] = (_Float16)v;
        }
    }

    const float fx = (float)x, fy = (float)y;
    float cwsum = 1e-8f;
    float cf[4] = {0.f, 0.f, 0.f, 0.f};

    #pragma unroll
    for (int s = 0; s < 4; ++s) {
        const float* M = RT + s * 12;
        float rx0 = M[0] * fx + M[1] * fy + M[2];
        float rx1 = M[3] * fx + M[4] * fy + M[5];
        float rx2 = M[6] * fx + M[7] * fy + M[8];
        float pxw = rx0 * depth + M[9];
        float pyw = rx1 * depth + M[10];
        float pzw = rx2 * depth + M[11];
        float iz = 1.0f / pzw;                 // exact: feeds tap selection
        float px = pxw * iz, py = pyw * iz;

        float x0f = floorf(px), y0f = floorf(py);
        float wx = px - x0f, wy = py - y0f;

        float warped[16];
        #pragma unroll
        for (int c = 0; c < 16; ++c) warped[c] = 0.0f;

        const float* sp = feat + (size_t)(s + 1) * Cc * HW;
        #pragma unroll
        for (int t = 0; t < 4; ++t) {
            float xi = x0f + (float)(t & 1);
            float yi = y0f + (float)(t >> 1);
            float w = ((t & 1) ? wx : 1.0f - wx) * ((t >> 1) ? wy : 1.0f - wy);
            bool valid = (xi >= 0.0f) && (xi <= (float)(Wc - 1)) &&
                         (yi >= 0.0f) && (yi <= (float)(Hc - 1));
            float we = valid ? w : 0.0f;
            int xc = (int)fminf(fmaxf(xi, 0.0f), (float)(Wc - 1));
            int yc = (int)fminf(fmaxf(yi, 0.0f), (float)(Hc - 1));
            const float* tp = sp + (size_t)yc * Wc + xc;
            #pragma unroll
            for (int c = 0; c < 16; ++c) warped[c] += we * tp[(size_t)c * HW];
        }

        // ---- pack warped to f16 B operands ----
        v16h Bm;
        #pragma unroll
        for (int c = 0; c < 16; ++c) Bm[c] = (_Float16)warped[c];

        v16h B1;   // depths 0..15 (columns = lanes 0..15), K>=16 lanes zero
        #pragma unroll
        for (int c = 0; c < 16; ++c) B1[c] = lo ? Bm[c] : (_Float16)0.0f;

        // depths 16..31 -> lanes 0..15 via SWAPX16 ds_swizzle (unconditional)
        v16h B2;
        {
            union { v16h h; int u[8]; } cvt; cvt.h = Bm;
            union { v16h h; int u[8]; } ov;
            #pragma unroll
            for (int j = 0; j < 8; ++j)
                ov.u[j] = __builtin_amdgcn_ds_swizzle(cvt.u[j], SWZ_X16);
            #pragma unroll
            for (int c = 0; c < 16; ++c) B2[c] = lo ? ov.h[c] : (_Float16)0.0f;
        }

        v8f zc = {0.f, 0.f, 0.f, 0.f, 0.f, 0.f, 0.f, 0.f};
        v8f D1 = __builtin_amdgcn_wmma_f32_16x16x32_f16(
            false, Amat, false, B1, (short)0, zc, false, false);
        v8f D2 = __builtin_amdgcn_wmma_f32_16x16x32_f16(
            false, Amat, false, B2, (short)0, zc, false, false);

        // cor(g, my_depth) in every lane; all swizzles consumed by all lanes.
        float cor[4];
        #pragma unroll
        for (int g = 0; g < 4; ++g) {
            float masked = lo ? D2[g] : 0.0f;       // D2 col for hi half
            cor[g] = D1[g] + FSWZ(masked, SWZ_X16); // D1 rows 4..15 are zero
        }

        // ---- view attention: softmax over D (full-wave swizzle tree) ----
        float tt = (cor[0] + cor[1] + cor[2] + cor[3]) * 0.5f;  // / ATTN_TEMP
        float m = wave_max32(tt);
        float e = __expf(tt - m);                 // arg <= 0: fast exp safe
        float Z = wave_sum32(e);
        float w = e * (0.25f * __builtin_amdgcn_rcpf(Z));  // softmax / sqrt(C)
        cwsum += w;
        #pragma unroll
        for (int g = 0; g < 4; ++g) cf[g] += w * cor[g];
    }

    // ---- cost regression + softmax over D ----
    float invs = __builtin_amdgcn_rcpf(cwsum);
    float creg = (cf[0]*rw[0] + cf[1]*rw[1] + cf[2]*rw[2] + cf[3]*rw[3]) * invs;
    float m2 = wave_max32(creg);
    float e2 = __expf(creg - m2);
    float Z2 = wave_sum32(e2);
    float p = e2 * __builtin_amdgcn_rcpf(Z2);

    const size_t pidx = (size_t)y * Wc + x;
    out[(size_t)2 * HW + (size_t)lane * HW + pidx] = p;   // prob_volume

    // ---- argmax (first max, matching jnp.argmax) + scalar outputs ----
    float pm = wave_max32(p);
    unsigned long long ball = __ballot(p == pm);
    int first = __ffsll(ball) - 1;                        // wave-uniform
    float dsel = read_lane_f(depth, first);

    float invd = 1.0f / depth;                            // exact: output path
    float litv = read_lane_f(invd, 2) - read_lane_f(invd, 1);  // 1/dh[2]-1/dh[1]
    float r = dir[0];

    if (lane == 0) {
        out[pidx] = dsel;                                  // depth
        out[(size_t)HW + pidx] = pm;                       // confidence
        out[(size_t)2 * HW + (size_t)Dc * HW + pidx] = 1.0f / dsel + r * litv;
        out[(size_t)2 * HW + (size_t)(Dc + 1) * HW + pidx] = 1.0f / dsel - r * litv;
    }
}

extern "C" void kernel_launch(void* const* d_in, const int* in_sizes, int n_in,
                              void* d_out, int out_size, void* d_ws, size_t ws_size,
                              hipStream_t stream) {
    const float* features = (const float*)d_in[0];
    const float* projs    = (const float*)d_in[1];
    const float* dh       = (const float*)d_in[2];
    const float* rw       = (const float*)d_in[3];
    const float* dir      = (const float*)d_in[4];
    float* out = (float*)d_out;
    float* ws  = (float*)d_ws;
    (void)in_sizes; (void)n_in; (void)out_size; (void)ws_size;

    proj_precompute_kernel<<<1, 32, 0, stream>>>(projs, ws);
    stagenet_kernel<<<HW / 8, 256, 0, stream>>>(features, dh, rw, dir, ws, out);
}
